// PosEmb_42305427866161
// MI455X (gfx1250) — matrix operand order repaired and checked by reading, not compiled, not run on hardware
//
#include <hip/hip_runtime.h>
#include <math.h>

typedef __attribute__((ext_vector_type(16))) _Float16     v16h;
typedef __attribute__((ext_vector_type(2)))  __fp16       v2fp16;
typedef __attribute__((ext_vector_type(8)))  float        v8f;
typedef __attribute__((ext_vector_type(4)))  unsigned int u32x4;

#define HW      3600
#define CDIM    128
#define WIDTH   60
#define NB      4
#define KTOP    100
#define SCALE   0.078125f     /* 1/(128*0.1) */
#define CANDMAX 8192

/* ---------------- workspace layout (bytes, 256-aligned) ---------------- */
static const size_t OFF_SM    = 0;                       /* 4*3600*3600 f32 = 207,360,000 */
static const size_t OFF_RMAX  = 207360000;               /* 4*3600 f32 */
static const size_t OFF_RSUM  = 207417600;
static const size_t OFF_CMAX  = 207475200;
static const size_t OFF_CSUM  = 207532800;
static const size_t OFF_HIST  = 207590400;               /* 4*65536 u32 = 1,048,576 */
static const size_t OFF_THR   = 208638976;               /* 4 u32 */
static const size_t OFF_CNT   = 208639232;               /* 4 u32 */
static const size_t OFF_CANDV = 208639488;               /* 4*8192 f32 */
static const size_t OFF_CANDI = 208770560;               /* 4*8192 i32 */
static const size_t OFF_TOPI  = 208901632;               /* 4*100 i32 */
static const size_t OFF_SXY   = 208903680;               /* 2*128 f32 */
static const size_t OFF_G     = 208904704;               /* 8*128 f32 */

/* one v_cvt_pkrtz_f16_f32: pack (a,b) into low/high halves of a DWORD */
__device__ __forceinline__ unsigned pk2h(float a, float b) {
    union { v2fp16 h; unsigned u; } p;
    p.h = __builtin_amdgcn_cvt_pkrtz(a, b);
    return p.u;
}

/* =======================================================================
 * K1: batched GEMM  sm[n][l][s] = (1/(c*T)) * sum_d x0[n,d,l]*x1[n,d,s]
 * 128x64 C tile per block, 8 waves x (32x32 C each, 4 accumulators),
 * v_wmma_f32_16x16x32_f16 inner op, b128 staging + packed b32 DS stores.
 * ===================================================================== */
__global__ __launch_bounds__(256) void gemm_sm_kernel(const float* __restrict__ x,
                                                      float* __restrict__ sm) {
    __shared__ _Float16 As[128][40];  /* [m][k], pad 40: bank-friendly */
    __shared__ _Float16 Bs[64][40];   /* [s][k] */

    const int n     = blockIdx.z;
    const int mBase = blockIdx.y * 128;
    const int nBase = blockIdx.x * 64;
    const float* __restrict__ A = x + (size_t)n       * CDIM * HW;  /* f0: [d][l] */
    const float* __restrict__ B = x + (size_t)(n + NB)* CDIM * HW;  /* f1: [d][s] */

    const int tid  = threadIdx.x;
    const int lane = tid & 31;
    const int wave = tid >> 5;
    const int wm   = (wave & 3) * 32;   /* wave M offset (0/32/64/96) */
    const int wn   = (wave >> 2) * 32;  /* wave N offset (0/32)       */

    /* staging ownership: 2 consecutive k x contiguous run of m (or n) */
    const int kp = (tid & 15) * 2;      /* k pair base: 0,2,..,30   */
    const int mg = (tid >> 4) * 8;      /* A: 8 m's: 0,8,..,120     */
    const int ng = (tid >> 4) * 4;      /* B: 4 n's: 0,4,..,60      */
    int gm0 = mBase + mg; if (gm0 > HW - 8) gm0 = HW - 8;   /* keep runs in-bounds;   */
    int gn0 = nBase + ng; if (gn0 > HW - 4) gn0 = HW - 4;   /* OOB tiles never stored */

    v8f acc00 = {}, acc01 = {}, acc10 = {}, acc11 = {};

    for (int ks = 0; ks < 4; ++ks) {
        const int k0 = ks * 32;
        /* ---- stage A(128x32) and B(64x32) tiles, f32 -> packed f16 ---- */
        {
            const float* sa0 = A + (size_t)(k0 + kp)     * HW;
            const float* sa1 = A + (size_t)(k0 + kp + 1) * HW;
            const float* sb0 = B + (size_t)(k0 + kp)     * HW;
            const float* sb1 = B + (size_t)(k0 + kp + 1) * HW;
            if (ks < 3) {   /* prefetch next K-slab (global_prefetch_b8) */
                __builtin_prefetch(sa0 + 32 * HW + gm0, 0, 3);
                __builtin_prefetch(sb0 + 32 * HW + gn0, 0, 3);
            }
            float ra0[8], ra1[8];
#pragma unroll
            for (int j = 0; j < 2; ++j) {
                *(float4*)&ra0[j * 4] = *(const float4*)(sa0 + gm0 + j * 4);
                *(float4*)&ra1[j * 4] = *(const float4*)(sa1 + gm0 + j * 4);
            }
#pragma unroll
            for (int j = 0; j < 8; ++j)
                *(unsigned*)&As[mg + j][kp] = pk2h(ra0[j], ra1[j]);

            float rb0[4], rb1[4];
            *(float4*)&rb0[0] = *(const float4*)(sb0 + gn0);
            *(float4*)&rb1[0] = *(const float4*)(sb1 + gn0);
#pragma unroll
            for (int j = 0; j < 4; ++j)
                *(unsigned*)&Bs[ng + j][kp] = pk2h(rb0[j], rb1[j]);
        }
        __syncthreads();

        /* ---- fragment loads per documented 16-bit WMMA VGPR layouts ---- */
        union Frag { v16h v; u32x4 q[2]; };
        Frag a0, a1, b0, b1;
        const int am    = wm + (lane & 15);
        const int abase = (lane >> 4) * 8;   /* A: lanes 0-15 K{0-7,16-23}, 16-31 K{8-15,24-31} */
        a0.q[0] = *(const u32x4*)&As[am][abase];
        a0.q[1] = *(const u32x4*)&As[am][abase + 16];
        a1.q[0] = *(const u32x4*)&As[am + 16][abase];
        a1.q[1] = *(const u32x4*)&As[am + 16][abase + 16];
        const int bn    = wn + (lane & 15);
        const int bbase = (lane >> 4) * 8;   /* B: lanes 0-15 K{0-7,16-23}, 16-31 K{8-15,24-31} */
        b0.q[0] = *(const u32x4*)&Bs[bn][bbase];
        b0.q[1] = *(const u32x4*)&Bs[bn][bbase + 16];
        b1.q[0] = *(const u32x4*)&Bs[bn + 16][bbase];
        b1.q[1] = *(const u32x4*)&Bs[bn + 16][bbase + 16];

        acc00 = __builtin_amdgcn_wmma_f32_16x16x32_f16(false, a0.v, false, b0.v,
                                                       (short)0, acc00, false, false);
        acc01 = __builtin_amdgcn_wmma_f32_16x16x32_f16(false, a0.v, false, b1.v,
                                                       (short)0, acc01, false, false);
        acc10 = __builtin_amdgcn_wmma_f32_16x16x32_f16(false, a1.v, false, b0.v,
                                                       (short)0, acc10, false, false);
        acc11 = __builtin_amdgcn_wmma_f32_16x16x32_f16(false, a1.v, false, b1.v,
                                                       (short)0, acc11, false, false);
        __syncthreads();
    }

    /* ---- store per C-layout: VGPR v -> M=v (lanes 0-15) / v+8 (16-31) ---- */
    float* __restrict__ C = sm + (size_t)n * HW * HW;
    const int colBase = nBase + wn + (lane & 15);
    const int rAdd    = (lane >> 4) * 8;
#pragma unroll
    for (int v = 0; v < 8; ++v) {
        const int r0 = mBase + wm + rAdd + v;
        const int r1 = r0 + 16;
        if (colBase < HW) {
            if (r0 < HW) C[(size_t)r0 * HW + colBase] = acc00[v] * SCALE;
            if (r1 < HW) C[(size_t)r1 * HW + colBase] = acc10[v] * SCALE;
        }
        if (colBase + 16 < HW) {
            if (r0 < HW) C[(size_t)r0 * HW + colBase + 16] = acc01[v] * SCALE;
            if (r1 < HW) C[(size_t)r1 * HW + colBase + 16] = acc11[v] * SCALE;
        }
    }
}

/* ============ K2a: per-row online softmax stats (max, sumexp) ============ */
__global__ __launch_bounds__(256) void row_stats_kernel(const float* __restrict__ sm,
                                                        float* __restrict__ rmax,
                                                        float* __restrict__ rsum) {
    const int l = blockIdx.x, n = blockIdx.y;
    const float* row = sm + ((size_t)n * HW + l) * HW;
    float m = -3.0e38f, d = 0.f;
    for (int s = threadIdx.x; s < HW; s += 256) {
        float v = row[s];
        if (v > m) { d = d * __expf(m - v) + 1.f; m = v; }
        else       { d += __expf(v - m); }
    }
    __shared__ float sM[256], sD[256];
    sM[threadIdx.x] = m; sD[threadIdx.x] = d;
    __syncthreads();
    for (int off = 128; off > 0; off >>= 1) {
        if (threadIdx.x < off) {
            float m1 = sM[threadIdx.x], d1 = sD[threadIdx.x];
            float m2 = sM[threadIdx.x + off], d2 = sD[threadIdx.x + off];
            float M = fmaxf(m1, m2);
            sM[threadIdx.x] = M;
            sD[threadIdx.x] = d1 * __expf(m1 - M) + d2 * __expf(m2 - M);
        }
        __syncthreads();
    }
    if (threadIdx.x == 0) { rmax[n * HW + l] = sM[0]; rsum[n * HW + l] = sD[0]; }
}

/* ====== K2b: per-column stats, thread-per-column (coalesced sweeps) ====== */
__global__ __launch_bounds__(256) void col_stats_kernel(const float* __restrict__ sm,
                                                        float* __restrict__ cmax,
                                                        float* __restrict__ csum) {
    const int s = blockIdx.x * 256 + threadIdx.x;
    const int n = blockIdx.y;
    if (s >= HW) return;
    const float* base = sm + (size_t)n * HW * HW + s;
    float m = -3.0e38f, d = 0.f;
    for (int l = 0; l < HW; ++l) {
        float v = base[(size_t)l * HW];
        if (v > m) { d = d * __expf(m - v) + 1.f; m = v; }
        else       { d += __expf(v - m); }
    }
    cmax[n * HW + s] = m; csum[n * HW + s] = d;
}

/* ================= K3z: zero histogram + counters ================= */
__global__ void zero_kernel(unsigned int* hist, unsigned int* cnt, unsigned int* thr) {
    int i = blockIdx.x * 256 + threadIdx.x;
    if (i < NB * 65536) hist[i] = 0u;
    if (i < NB) { cnt[i] = 0u; thr[i] = 0u; }
}

__device__ __forceinline__ float cm_value(const float* S, const float* rm, const float* rs,
                                          const float* cx, const float* cs, int i) {
    int l = i / HW, s = i - l * HW;
    return __expf(2.f * S[i] - rm[l] - cx[s]) / (rs[l] * cs[s]);
}

/* ====== K3a: 16-bit float-ordered-key histogram of cm (v in [0,1]) ====== */
__global__ __launch_bounds__(256) void cm_hist_kernel(const float* __restrict__ sm,
        const float* __restrict__ rmax, const float* __restrict__ rsum,
        const float* __restrict__ cmax, const float* __restrict__ csum,
        unsigned int* __restrict__ hist) {
    const int n = blockIdx.y;
    const float* S  = sm   + (size_t)n * HW * HW;
    const float* rm = rmax + n * HW; const float* rs = rsum + n * HW;
    const float* cx = cmax + n * HW; const float* cs = csum + n * HW;
    const int total = HW * HW, stride = gridDim.x * 256;
    for (int i = blockIdx.x * 256 + threadIdx.x; i < total; i += stride) {
        float v = cm_value(S, rm, rs, cx, cs, i);
        unsigned key = __float_as_uint(v) >> 15;   /* monotonic for v >= 0 */
        atomicAdd(&hist[((unsigned)n << 16) + key], 1u);
    }
}

/* ====== K3b: descending scan -> threshold bucket holding >= K elems ====== */
__global__ void thresh_kernel(const unsigned int* __restrict__ hist, unsigned int* thr) {
    const int n = blockIdx.x;
    if (threadIdx.x == 0) {
        unsigned c = 0; int k = 65535;
        for (; k > 0; --k) { c += hist[((unsigned)n << 16) + k]; if (c >= KTOP) break; }
        thr[n] = (unsigned)k;
    }
}

/* ====== K3c: collect candidates with key >= threshold ====== */
__global__ __launch_bounds__(256) void collect_kernel(const float* __restrict__ sm,
        const float* __restrict__ rmax, const float* __restrict__ rsum,
        const float* __restrict__ cmax, const float* __restrict__ csum,
        const unsigned int* __restrict__ thr, unsigned int* __restrict__ cnt,
        float* __restrict__ candV, int* __restrict__ candI) {
    const int n = blockIdx.y;
    const float* S  = sm   + (size_t)n * HW * HW;
    const float* rm = rmax + n * HW; const float* rs = rsum + n * HW;
    const float* cx = cmax + n * HW; const float* cs = csum + n * HW;
    const unsigned t = thr[n];
    const int total = HW * HW, stride = gridDim.x * 256;
    for (int i = blockIdx.x * 256 + threadIdx.x; i < total; i += stride) {
        float v = cm_value(S, rm, rs, cx, cs, i);
        if ((__float_as_uint(v) >> 15) >= t) {
            unsigned pos = atomicAdd(&cnt[n], 1u);
            if (pos < CANDMAX) { candV[n * CANDMAX + pos] = v; candI[n * CANDMAX + pos] = i; }
        }
    }
}

/* ====== K3d: exact top-K over candidates (one block per batch) ====== */
__global__ __launch_bounds__(256) void topk_select_kernel(float* __restrict__ candV,
        const int* __restrict__ candI, const unsigned int* __restrict__ cnt,
        int* __restrict__ topidx) {
    const int n = blockIdx.x;
    unsigned m = cnt[n]; if (m > CANDMAX) m = CANDMAX;
    __shared__ float bV[256]; __shared__ int bI[256];
    for (int k = 0; k < KTOP; ++k) {
        float best = -1.f; int bi = -1;
        for (int i = threadIdx.x; i < (int)m; i += 256) {
            float v = candV[n * CANDMAX + i];
            if (v > best) { best = v; bi = i; }
        }
        bV[threadIdx.x] = best; bI[threadIdx.x] = bi;
        __syncthreads();
        for (int off = 128; off > 0; off >>= 1) {
            if (threadIdx.x < off && bV[threadIdx.x + off] > bV[threadIdx.x]) {
                bV[threadIdx.x] = bV[threadIdx.x + off];
                bI[threadIdx.x] = bI[threadIdx.x + off];
            }
            __syncthreads();
        }
        if (threadIdx.x == 0) {
            int w = bI[0];
            topidx[n * KTOP + k] = (w >= 0) ? candI[n * CANDMAX + w] : 0;
            if (w >= 0) candV[n * CANDMAX + w] = -2.f;
        }
        __syncthreads();
    }
}

/* ====== K4a: embedding collapses to rank-2 update + per-batch constant ======
 * emb[l,d] = gx[l]*Sx[d] + gy[l]*Sy[d] + (b[d] - T[n,d])                     */
__global__ void emb_const_kernel(const float* __restrict__ W, const float* __restrict__ b,
                                 const int* __restrict__ topidx,
                                 float* __restrict__ Sxy, float* __restrict__ G) {
    const int d = threadIdx.x;            /* 128 threads */
    const float* w = W + d * 2 * KTOP;
    float sx = 0.f, sy = 0.f;
    for (int k = 0; k < KTOP; ++k) { sx += w[2 * k]; sy += w[2 * k + 1]; }
    Sxy[d] = sx; Sxy[CDIM + d] = sy;
    const float inv = 1.f / (float)WIDTH;
    for (int n = 0; n < NB; ++n) {
        float t0 = 0.f, t1 = 0.f;
        for (int k = 0; k < KTOP; ++k) {
            int idx = topidx[n * KTOP + k];
            int q = idx / HW, r = idx - q * HW;
            t0 += w[2 * k] * ((q % WIDTH) * inv) + w[2 * k + 1] * ((q / WIDTH) * inv);
            t1 += w[2 * k] * ((r % WIDTH) * inv) + w[2 * k + 1] * ((r / WIDTH) * inv);
        }
        G[n * CDIM + d]        = b[d] - t0;
        G[(n + NB) * CDIM + d] = b[d] - t1;
    }
}

/* ====== K4b: out[i,d,l] = x[i,d,l] + gx*Sx[d] + gy*Sy[d] + G[i,d] ====== */
__global__ __launch_bounds__(256) void output_kernel(const float* __restrict__ x,
        const float* __restrict__ Sxy, const float* __restrict__ G,
        float* __restrict__ out) {
    const size_t total = (size_t)2 * NB * CDIM * HW;
    size_t i = (size_t)blockIdx.x * 256 + threadIdx.x;
    if (i >= total) return;
    int l  = (int)(i % HW);
    int d  = (int)((i / HW) % CDIM);
    int bt = (int)(i / ((size_t)HW * CDIM));
    const float inv = 1.f / (float)WIDTH;
    float gx = (l % WIDTH) * inv, gy = (l / WIDTH) * inv;
    out[i] = x[i] + gx * Sxy[d] + gy * Sxy[CDIM + d] + G[bt * CDIM + d];
}

/* ======================================================================= */
extern "C" void kernel_launch(void* const* d_in, const int* in_sizes, int n_in,
                              void* d_out, int out_size, void* d_ws, size_t ws_size,
                              hipStream_t stream) {
    (void)in_sizes; (void)n_in; (void)out_size; (void)ws_size;
    const float* x = (const float*)d_in[0];
    const float* W = (const float*)d_in[1];
    const float* b = (const float*)d_in[2];
    float* out = (float*)d_out;

    char* ws = (char*)d_ws;
    float*        sm    = (float*)(ws + OFF_SM);
    float*        rmax  = (float*)(ws + OFF_RMAX);
    float*        rsum  = (float*)(ws + OFF_RSUM);
    float*        cmax  = (float*)(ws + OFF_CMAX);
    float*        csum  = (float*)(ws + OFF_CSUM);
    unsigned int* hist  = (unsigned int*)(ws + OFF_HIST);
    unsigned int* thr   = (unsigned int*)(ws + OFF_THR);
    unsigned int* cnt   = (unsigned int*)(ws + OFF_CNT);
    float*        candV = (float*)(ws + OFF_CANDV);
    int*          candI = (int*)(ws + OFF_CANDI);
    int*          topi  = (int*)(ws + OFF_TOPI);
    float*        Sxy   = (float*)(ws + OFF_SXY);
    float*        G     = (float*)(ws + OFF_G);

    const int mt = (HW + 127) / 128;  /* 29 M-tiles */
    const int nt = (HW + 63) / 64;    /* 57 N-tiles */
    gemm_sm_kernel<<<dim3(nt, mt, NB), 256, 0, stream>>>(x, sm);
    row_stats_kernel<<<dim3(HW, NB), 256, 0, stream>>>(sm, rmax, rsum);
    col_stats_kernel<<<dim3((HW + 255) / 256, NB), 256, 0, stream>>>(sm, cmax, csum);
    zero_kernel<<<(NB * 65536 + 255) / 256, 256, 0, stream>>>(hist, cnt, thr);
    cm_hist_kernel<<<dim3(512, NB), 256, 0, stream>>>(sm, rmax, rsum, cmax, csum, hist);
    thresh_kernel<<<NB, 32, 0, stream>>>(hist, thr);
    collect_kernel<<<dim3(512, NB), 256, 0, stream>>>(sm, rmax, rsum, cmax, csum,
                                                      thr, cnt, candV, candI);
    topk_select_kernel<<<NB, 256, 0, stream>>>(candV, candI, cnt, topi);
    emb_const_kernel<<<1, CDIM, 0, stream>>>(W, b, topi, Sxy, G);
    output_kernel<<<(int)(((size_t)2 * NB * CDIM * HW + 255) / 256), 256, 0, stream>>>(x, Sxy, G, out);
}
